// MoETransformerEncoderLayer_71451075936555
// MI455X (gfx1250) — compile-verified, matrix operand order checked by reading, not run on hardware
//
#include <hip/hip_runtime.h>
#include <hip/hip_bf16.h>
#include <math.h>

// ---------------------------------------------------------------------------
// MoE transformer encoder layer for MI455X (gfx1250), wave32 + WMMA bf16.
// GEMMs: v_wmma_f32_16x16x32_bf16, double-buffered LDS with
// global_load_async_to_lds_b128 staging (ASYNCcnt).
// Attention: WMMA flash kernel with online softmax.
// ---------------------------------------------------------------------------

typedef __bf16 bf16_t;
typedef __bf16 bf16x8 __attribute__((ext_vector_type(8)));
typedef __bf16 v16bf  __attribute__((ext_vector_type(16)));
typedef float  v8f    __attribute__((ext_vector_type(8)));

#define S_TOK   4096
#define DIM     1024
#define NH      16
#define DHD     64
#define DCMP    256
#define HIDN    2048
#define NEXP    8
#define BB      2
#define TT      2048

// --------------------- gfx1250 async-to-LDS helpers ------------------------
// Builtin signature (from clang diagnostic): first arg is
//   int __vector(4) addrspace(1)*  (printed as "__vector_size__(16) int __device__*")
// so pass AS(1)/AS(3) pointers to gcc-style int4 vectors.
typedef int v4i_t __attribute__((vector_size(16)));
typedef __attribute__((address_space(1))) v4i_t as1_v4i;
typedef __attribute__((address_space(3))) v4i_t as3_v4i;

#if __has_builtin(__builtin_amdgcn_global_load_async_to_lds_b128)
#define USE_ASYNC_LDS 1
#else
#define USE_ASYNC_LDS 0
#endif

// 16B per-lane copy: LDS[l] = MEM[g] tracked by ASYNCcnt (ISA 08 §4.4).
__device__ __forceinline__ void async_copy16(void* l, const void* g) {
#if USE_ASYNC_LDS
  __builtin_amdgcn_global_load_async_to_lds_b128((as1_v4i*)g, (as3_v4i*)l, 0, 0);
#else
  *(bf16x8*)l = *(const bf16x8*)g;
#endif
}

__device__ __forceinline__ void async_wait() {
#if USE_ASYNC_LDS
#if __has_builtin(__builtin_amdgcn_s_wait_asynccnt)
  __builtin_amdgcn_s_wait_asynccnt(0);
#else
  asm volatile("s_wait_asynccnt 0x0" ::: "memory");
#endif
#endif
}

// ---------------------------------------------------------------------------
__device__ __forceinline__ v8f vzero8() {
  v8f v;
#pragma unroll
  for (int i = 0; i < 8; ++i) v[i] = 0.f;
  return v;
}

// Load a 16x32 bf16 WMMA A/B fragment from a row-major tile (ldk stride,
// 16B-aligned rows). Lane layout per CDNA5 ISA 7.12.2:
//   lane<16 : row=lane,    K = {0..7, 16..23}
//   lane>=16: row=lane-16, K = {8..15, 24..31}
__device__ __forceinline__ v16bf ld_frag(const bf16_t* __restrict__ base, int ldk, int lane) {
  const int r  = lane & 15;
  const int h8 = (lane >> 4) << 3;           // 0 or 8
  const bf16_t* p = base + r * ldk + h8;
  bf16x8 lo = *(const bf16x8*)(p);
  bf16x8 hi = *(const bf16x8*)(p + 16);
  return __builtin_shufflevector(lo, hi, 0,1,2,3,4,5,6,7,8,9,10,11,12,13,14,15);
}

__device__ __forceinline__ v8f wmma_bf16(v16bf a, v16bf b, v8f c) {
  return __builtin_amdgcn_wmma_f32_16x16x32_bf16(
      /*neg_a=*/false, a, /*neg_b=*/false, b,
      /*c_mod=*/(short)0, c, /*reuse_a=*/false, /*reuse_b=*/false);
}

// ---------------------------------------------------------------------------
// Generic bf16 GEMM: C(MxN) = A(MxK) @ B(KxN). 256 threads (8 waves),
// block tile 128x128, K-step 32, wave tile 64x32 = 4x2 WMMA tiles.
// Double-buffered LDS: A tile staged with async-to-LDS copies, B tile loaded
// to VGPRs before the WMMA block and scattered (transposed) after it, so
// global latency overlaps matrix math. SCALED variant: C += (A@B)*w[row],
// skipping 128-row blocks whose routing weights are all zero.
// ---------------------------------------------------------------------------
#define GBM 128
#define GBN 128
#define GBK 32
#define GLD 40   // padded LDS K-stride (80B rows keep 16B alignment)

template<bool OUT_BF16, bool SCALED>
__global__ __launch_bounds__(256)
void gemm_bf16_kernel(const bf16_t* __restrict__ A, const bf16_t* __restrict__ B,
                      void* __restrict__ C, const float* __restrict__ rowscale,
                      int rs_stride, int M, int N, int K)
{
  __shared__ alignas(16) bf16_t As[2][GBM * GLD];
  __shared__ alignas(16) bf16_t Bs[2][GBN * GLD];   // transposed: [n][k]
  __shared__ int s_any;

  const int tid = threadIdx.x;
  const int m0  = blockIdx.y * GBM;
  const int n0  = blockIdx.x * GBN;

  if (SCALED) {
    if (tid == 0) s_any = 0;
    __syncthreads();
    if (tid < GBM && rowscale[(size_t)(m0 + tid) * rs_stride] != 0.f) s_any = 1;
    __syncthreads();
    if (!s_any) return;
  }

  const int lane = tid & 31;
  const int wv   = tid >> 5;
  const int wrow = (wv & 1) * 64;
  const int wcol = (wv >> 1) * 32;

  v8f acc[4][2];
#pragma unroll
  for (int mi = 0; mi < 4; ++mi)
#pragma unroll
    for (int ni = 0; ni < 2; ++ni) acc[mi][ni] = vzero8();

  const int nk = K / GBK;

  // ---- prologue: fully stage tile 0 into buffer 0 ----
#pragma unroll
  for (int i = 0; i < 2; ++i) {
    int c = tid + 256 * i;
    int row = c >> 2, kc = (c & 3) * 8;
    async_copy16(&As[0][row * GLD + kc], &A[(size_t)(m0 + row) * K + kc]);
  }
#pragma unroll
  for (int i = 0; i < 2; ++i) {
    int c = tid + 256 * i;
    int kb = c >> 4, nc = (c & 15) * 8;
    bf16x8 vb = *(const bf16x8*)&B[(size_t)kb * N + n0 + nc];
#pragma unroll
    for (int j = 0; j < 8; ++j) Bs[0][(nc + j) * GLD + kb] = vb[j];
  }

  for (int it = 0; it < nk; ++it) {
    const int cur = it & 1;
    async_wait();        // my async copies into As[cur] are complete
    __syncthreads();     // everyone's staging (async + ds scatter) visible

    // ---- issue next tile's loads before compute (latency hiding) ----
    bf16x8 vb0, vb1;
    const bool have_next = (it + 1) < nk;
    if (have_next) {
      const int kk = (it + 1) * GBK;
#pragma unroll
      for (int i = 0; i < 2; ++i) {
        int c = tid + 256 * i;
        int row = c >> 2, kc = (c & 3) * 8;
        async_copy16(&As[cur ^ 1][row * GLD + kc],
                     &A[(size_t)(m0 + row) * K + kk + kc]);
      }
      {
        int c = tid;
        int kb = c >> 4, nc = (c & 15) * 8;
        vb0 = *(const bf16x8*)&B[(size_t)(kk + kb) * N + n0 + nc];
      }
      {
        int c = tid + 256;
        int kb = c >> 4, nc = (c & 15) * 8;
        vb1 = *(const bf16x8*)&B[(size_t)(kk + kb) * N + n0 + nc];
      }
    }

    // ---- WMMA block on buffer `cur` ----
    v16bf bf[2];
#pragma unroll
    for (int ni = 0; ni < 2; ++ni)
      bf[ni] = ld_frag(&Bs[cur][(wcol + ni * 16) * GLD], GLD, lane);
#pragma unroll
    for (int mi = 0; mi < 4; ++mi) {
      v16bf af = ld_frag(&As[cur][(wrow + mi * 16) * GLD], GLD, lane);
#pragma unroll
      for (int ni = 0; ni < 2; ++ni)
        acc[mi][ni] = wmma_bf16(af, bf[ni], acc[mi][ni]);
    }

    // ---- scatter next B tile (transpose) into the other buffer ----
    if (have_next) {
      {
        int c = tid;
        int kb = c >> 4, nc = (c & 15) * 8;
#pragma unroll
        for (int j = 0; j < 8; ++j) Bs[cur ^ 1][(nc + j) * GLD + kb] = vb0[j];
      }
      {
        int c = tid + 256;
        int kb = c >> 4, nc = (c & 15) * 8;
#pragma unroll
        for (int j = 0; j < 8; ++j) Bs[cur ^ 1][(nc + j) * GLD + kb] = vb1[j];
      }
    }
  }

  // ---- epilogue: C/D layout lane mapping (ISA 7.12.2) ----
  const int hlf = lane >> 4, nl = lane & 15;
#pragma unroll
  for (int mi = 0; mi < 4; ++mi) {
#pragma unroll
    for (int r = 0; r < 8; ++r) {
      const int row = m0 + wrow + mi * 16 + r + 8 * hlf;
      float sc = 1.f;
      if (SCALED) sc = rowscale[(size_t)row * rs_stride];
#pragma unroll
      for (int ni = 0; ni < 2; ++ni) {
        const int col = n0 + wcol + ni * 16 + nl;
        float v = acc[mi][ni][r];
        if (SCALED) {
          if (sc != 0.f) ((float*)C)[(size_t)row * N + col] += v * sc;
        } else if (OUT_BF16) {
          ((bf16_t*)C)[(size_t)row * N + col] = (bf16_t)v;
        } else {
          ((float*)C)[(size_t)row * N + col] = v;
        }
      }
    }
  }
}

// ---------------------------------------------------------------------------
// Flash attention (non-causal), head dim 64. Q/K/V in bf16 [B][H][T][64].
// 128 threads = 4 waves; each wave owns 16 query rows; 32-key LDS tiles.
// K tile staged with async-to-LDS; V transposed through VGPRs.
// Output written to [B][T][H*64] bf16 (ready for @Wo GEMM).
// ---------------------------------------------------------------------------
#define FLDK 72   // K-tile LDS stride   (144B rows, 16B aligned)
#define FLDP 40   // P / V^T LDS stride  (80B rows, 16B aligned)

__global__ __launch_bounds__(128)
void flash_attn_kernel(const bf16_t* __restrict__ Q, const bf16_t* __restrict__ K,
                       const bf16_t* __restrict__ V, bf16_t* __restrict__ O)
{
  __shared__ alignas(16) bf16_t Ks[32 * FLDK];
  __shared__ alignas(16) bf16_t Vs[64 * FLDP];      // transposed [d][key]
  __shared__ alignas(16) bf16_t Ps[4][16 * FLDP];   // per-wave P tile

  const int tid  = threadIdx.x;
  const int lane = tid & 31;
  const int wv   = tid >> 5;
  const int nl   = lane & 15, hlf = lane >> 4;
  const int bh   = blockIdx.y;
  const size_t off = (size_t)bh * TT * DHD;
  const int t0   = blockIdx.x * 64 + wv * 16;

  const v16bf qf0 = ld_frag(Q + off + (size_t)t0 * DHD,      DHD, lane);
  const v16bf qf1 = ld_frag(Q + off + (size_t)t0 * DHD + 32, DHD, lane);

  v8f o0 = vzero8(), o1 = vzero8(), o2 = vzero8(), o3 = vzero8();
  float mrow[8], lrow[8];
#pragma unroll
  for (int r = 0; r < 8; ++r) { mrow[r] = -1e30f; lrow[r] = 0.f; }

  for (int kb = 0; kb < TT; kb += 32) {
    // ---- stage K (async, row-major) and V (VGPR transpose) tiles ----
    bf16x8 vv[2];
#pragma unroll
    for (int i = 0; i < 2; ++i) {
      int c   = tid + 128 * i;
      int key = c >> 3;
      int dc  = (c & 7) * 8;
      async_copy16(&Ks[key * FLDK + dc], &K[off + (size_t)(kb + key) * DHD + dc]);
      vv[i] = *(const bf16x8*)&V[off + (size_t)(kb + key) * DHD + dc];
    }
#pragma unroll
    for (int i = 0; i < 2; ++i) {
      int c   = tid + 128 * i;
      int key = c >> 3;
      int dc  = (c & 7) * 8;
#pragma unroll
      for (int j = 0; j < 8; ++j) Vs[(dc + j) * FLDP + key] = vv[i][j];
    }
    async_wait();
    __syncthreads();

    // ---- S = Q @ K^T for two 16-key groups (K dim 64 -> 2 WMMAs each) ----
    v8f s0 = vzero8(), s1 = vzero8();
    s0 = wmma_bf16(qf0, ld_frag(Ks,                  FLDK, lane), s0);
    s0 = wmma_bf16(qf1, ld_frag(Ks + 32,             FLDK, lane), s0);
    s1 = wmma_bf16(qf0, ld_frag(Ks + 16 * FLDK,      FLDK, lane), s1);
    s1 = wmma_bf16(qf1, ld_frag(Ks + 16 * FLDK + 32, FLDK, lane), s1);

    // ---- online softmax (row stats replicated across 16 lanes) ----
#pragma unroll
    for (int r = 0; r < 8; ++r) {
      float a0 = s0[r] * 0.125f;   // 1/sqrt(64)
      float a1 = s1[r] * 0.125f;
      float mx = fmaxf(a0, a1);
#pragma unroll
      for (int d = 1; d < 16; d <<= 1) mx = fmaxf(mx, __shfl_xor(mx, d));
      float nm = fmaxf(mrow[r], mx);
      float e0 = __expf(a0 - nm), e1 = __expf(a1 - nm);
      float ps = e0 + e1;
#pragma unroll
      for (int d = 1; d < 16; d <<= 1) ps += __shfl_xor(ps, d);
      float corr = __expf(mrow[r] - nm);
      lrow[r] = lrow[r] * corr + ps;
      mrow[r] = nm;
      o0[r] *= corr; o1[r] *= corr; o2[r] *= corr; o3[r] *= corr;
      const int prow = r + 8 * hlf;     // C-layout row for this lane/reg
      Ps[wv][prow * FLDP + nl]      = (bf16_t)e0;
      Ps[wv][prow * FLDP + 16 + nl] = (bf16_t)e1;
    }
    __syncthreads();   // make P visible cross-lane, Vs ready

    // ---- O += P(16x32) @ V(32x64) : 4 WMMAs ----
    v16bf pf = ld_frag(Ps[wv], FLDP, lane);
    o0 = wmma_bf16(pf, ld_frag(Vs +  0 * FLDP, FLDP, lane), o0);
    o1 = wmma_bf16(pf, ld_frag(Vs + 16 * FLDP, FLDP, lane), o1);
    o2 = wmma_bf16(pf, ld_frag(Vs + 32 * FLDP, FLDP, lane), o2);
    o3 = wmma_bf16(pf, ld_frag(Vs + 48 * FLDP, FLDP, lane), o3);
    __syncthreads();   // before next tile overwrites Ks/Vs
  }

  const int b = bh / NH, h = bh % NH;
#pragma unroll
  for (int r = 0; r < 8; ++r) {
    const int t = t0 + r + 8 * hlf;
    const float inv = 1.f / lrow[r];
    const size_t ob = ((size_t)(b * TT + t)) * DIM + h * DHD;
    O[ob +  0 + nl] = (bf16_t)(o0[r] * inv);
    O[ob + 16 + nl] = (bf16_t)(o1[r] * inv);
    O[ob + 32 + nl] = (bf16_t)(o2[r] * inv);
    O[ob + 48 + nl] = (bf16_t)(o3[r] * inv);
  }
}

// ---------------------------------------------------------------------------
// RoPE on q,k + bf16 transpose to [B][H][T][64]; v just transposed/converted.
// ---------------------------------------------------------------------------
__global__ __launch_bounds__(32)
void rope_qkv_kernel(const float* __restrict__ q, const float* __restrict__ k,
                     const float* __restrict__ v, bf16_t* __restrict__ qT,
                     bf16_t* __restrict__ kT, bf16_t* __restrict__ vT)
{
  const int idx = blockIdx.x;          // (b*T + t)*H + h
  const int h  = idx % NH;
  const int bt = idx / NH;
  const int t  = bt % TT;
  const int b  = bt / TT;
  const int i  = threadIdx.x;          // pair index 0..31

  const float freq = __powf(10000.f, -(float)(2 * i) / (float)DHD);
  float sn, cs;
  __sincosf((float)t * freq, &sn, &cs);

  const size_t in_off  = (size_t)bt * DIM + h * DHD;
  const size_t out_off = (((size_t)(b * NH + h)) * TT + t) * DHD;

  float q1 = q[in_off + 2 * i], q2 = q[in_off + 2 * i + 1];
  qT[out_off + 2 * i]     = (bf16_t)(q1 * cs - q2 * sn);
  qT[out_off + 2 * i + 1] = (bf16_t)(q1 * sn + q2 * cs);

  float k1 = k[in_off + 2 * i], k2 = k[in_off + 2 * i + 1];
  kT[out_off + 2 * i]     = (bf16_t)(k1 * cs - k2 * sn);
  kT[out_off + 2 * i + 1] = (bf16_t)(k1 * sn + k2 * cs);

  vT[out_off + 2 * i]     = (bf16_t)v[in_off + 2 * i];
  vT[out_off + 2 * i + 1] = (bf16_t)v[in_off + 2 * i + 1];
}

// ---------------------------------------------------------------------------
// Fused residual add (2 or 3 way) + RMSNorm; optional fp32 and bf16 outputs.
// ---------------------------------------------------------------------------
__global__ __launch_bounds__(256)
void add_rmsnorm_kernel(const float* __restrict__ a, const float* __restrict__ b,
                        const float* __restrict__ c, const float* __restrict__ w,
                        float* __restrict__ outf, bf16_t* __restrict__ outb)
{
  const int row = blockIdx.x, tid = threadIdx.x;
  const size_t base = (size_t)row * DIM;
  __shared__ float red[256];
  float vals[4], ss = 0.f;
#pragma unroll
  for (int i = 0; i < 4; ++i) {
    const int j = tid + i * 256;
    float v = a[base + j];
    if (b) v += b[base + j];
    if (c) v += c[base + j];
    vals[i] = v;
    ss += v * v;
  }
  red[tid] = ss;
  __syncthreads();
  for (int st = 128; st > 0; st >>= 1) {
    if (tid < st) red[tid] += red[tid + st];
    __syncthreads();
  }
  const float inv = rsqrtf(red[0] / (float)DIM + 1e-6f);
#pragma unroll
  for (int i = 0; i < 4; ++i) {
    const int j = tid + i * 256;
    const float r = vals[i] * inv * w[j];
    if (outf) outf[base + j] = r;
    if (outb) outb[base + j] = (bf16_t)r;
  }
}

// ---------------------------------------------------------------------------
// MoE gating: one block per token. 10 logits (2 group + 8 expert), group
// argmax, in-group softmax * group prob, top-2 -> dense weights w[s][8].
// ---------------------------------------------------------------------------
__global__ __launch_bounds__(128)
void gate_kernel(const float* __restrict__ x, const float* __restrict__ Wg,
                 const float* __restrict__ We, const float* __restrict__ gbias,
                 const float* __restrict__ ebias, float* __restrict__ wout)
{
  const int s = blockIdx.x, tid = threadIdx.x;
  __shared__ float red[128];
  __shared__ float logit[10];
  const float* xr = x + (size_t)s * DIM;

  for (int li = 0; li < 10; ++li) {
    float p = 0.f;
    if (li < 2) {
      for (int j = tid; j < DIM; j += 128) p += xr[j] * Wg[j * 2 + li];
    } else {
      const int e = li - 2;
      for (int j = tid; j < DIM; j += 128) p += xr[j] * We[j * NEXP + e];
    }
    red[tid] = p;
    __syncthreads();
    for (int st = 64; st > 0; st >>= 1) {
      if (tid < st) red[tid] += red[tid + st];
      __syncthreads();
    }
    if (tid == 0) logit[li] = red[0];
    __syncthreads();
  }

  if (tid == 0) {
    const float g0 = logit[0] + gbias[0], g1 = logit[1] + gbias[1];
    const float gm = fmaxf(g0, g1);
    const float eg0 = __expf(g0 - gm), eg1 = __expf(g1 - gm);
    const int gidx = (g1 > g0) ? 1 : 0;
    const float gprob = (gidx ? eg1 : eg0) / (eg0 + eg1);

    float el[NEXP], pe[NEXP];
    float emax = -1e30f;
    for (int e = 0; e < NEXP; ++e) {
      el[e] = logit[2 + e] + ebias[e];
      if ((e >> 2) != gidx) el[e] = -1e30f;
      else emax = fmaxf(emax, el[e]);
    }
    float esum = 0.f;
    for (int e = 0; e < NEXP; ++e) {
      pe[e] = (el[e] <= -1e29f) ? 0.f : __expf(el[e] - emax);
      esum += pe[e];
    }
    for (int e = 0; e < NEXP; ++e) pe[e] = pe[e] / esum * gprob;

    int i1 = 0; float v1 = -1.f;
    for (int e = 0; e < NEXP; ++e) if (pe[e] > v1) { v1 = pe[e]; i1 = e; }
    int i2 = -1; float v2 = -1.f;
    for (int e = 0; e < NEXP; ++e) if (e != i1 && pe[e] > v2) { v2 = pe[e]; i2 = e; }

    for (int e = 0; e < NEXP; ++e) {
      float we = 0.f;
      if (e == i1) we = v1;
      else if (e == i2) we = v2;
      wout[(size_t)s * NEXP + e] = we;
    }
  }
}

// ---------------------------------------------------------------------------
// Elementwise helpers
// ---------------------------------------------------------------------------
__global__ __launch_bounds__(256)
void f32_to_bf16_kernel(const float* __restrict__ in, bf16_t* __restrict__ o, size_t n) {
  for (size_t i = blockIdx.x * 256ull + threadIdx.x; i < n; i += 256ull * gridDim.x)
    o[i] = (bf16_t)in[i];
}

__global__ __launch_bounds__(256)
void fill_zero_kernel(float* __restrict__ p, size_t n) {
  for (size_t i = blockIdx.x * 256ull + threadIdx.x; i < n; i += 256ull * gridDim.x)
    p[i] = 0.f;
}

__global__ __launch_bounds__(256)
void swiglu_kernel(const float* __restrict__ mid, bf16_t* __restrict__ o, size_t n) {
  for (size_t i = blockIdx.x * 256ull + threadIdx.x; i < n; i += 256ull * gridDim.x) {
    const size_t s = i / HIDN, j = i % HIDN;
    const float a = mid[s * (2 * HIDN) + j];
    const float b = mid[s * (2 * HIDN) + HIDN + j];
    const float sig = 1.f / (1.f + __expf(-a));
    o[i] = (bf16_t)(a * sig * b);
  }
}

// ---------------------------------------------------------------------------
// Host orchestration
// ---------------------------------------------------------------------------
extern "C" void kernel_launch(void* const* d_in, const int* in_sizes, int n_in,
                              void* d_out, int out_size, void* d_ws, size_t ws_size,
                              hipStream_t stream) {
  (void)in_sizes; (void)n_in; (void)out_size; (void)ws_size;

  const float* src   = (const float*)d_in[0];
  const float* Wq    = (const float*)d_in[1];
  const float* Wk_c  = (const float*)d_in[2];
  const float* Wv_c  = (const float*)d_in[3];
  const float* Wk    = (const float*)d_in[4];
  const float* Wv    = (const float*)d_in[5];
  const float* Wo    = (const float*)d_in[6];
  const float* Wsi   = (const float*)d_in[7];
  const float* Wso   = (const float*)d_in[8];
  const float* W1    = (const float*)d_in[9];
  const float* W2    = (const float*)d_in[10];
  const float* Wg    = (const float*)d_in[11];
  const float* We    = (const float*)d_in[12];
  const float* gbias = (const float*)d_in[13];
  const float* ebias = (const float*)d_in[14];
  const float* n1w   = (const float*)d_in[15];
  const float* n2w   = (const float*)d_in[16];
  float* out = (float*)d_out;

  char* ws = (char*)d_ws;
  size_t off = 0;
  auto alloc = [&](size_t bytes) -> char* {
    char* p = ws + off;
    off += (bytes + 255) & ~(size_t)255;
    return p;
  };

  // bf16 weights
  bf16_t* Wq16  = (bf16_t*)alloc((size_t)DIM * DIM * 2);
  bf16_t* Wkc16 = (bf16_t*)alloc((size_t)DIM * DCMP * 2);
  bf16_t* Wvc16 = (bf16_t*)alloc((size_t)DIM * DCMP * 2);
  bf16_t* Wk16  = (bf16_t*)alloc((size_t)DCMP * DIM * 2);
  bf16_t* Wv16  = (bf16_t*)alloc((size_t)DCMP * DIM * 2);
  bf16_t* Wo16  = (bf16_t*)alloc((size_t)DIM * DIM * 2);
  bf16_t* Wsi16 = (bf16_t*)alloc((size_t)DIM * 2 * HIDN * 2);
  bf16_t* Wso16 = (bf16_t*)alloc((size_t)HIDN * DIM * 2);
  bf16_t* W116  = (bf16_t*)alloc((size_t)DIM * 2 * HIDN * 2);
  bf16_t* W216  = (bf16_t*)alloc((size_t)NEXP * HIDN * DIM * 2);
  // activations
  bf16_t* src16 = (bf16_t*)alloc((size_t)S_TOK * DIM * 2);
  float*  qf    = (float*)alloc((size_t)S_TOK * DIM * 4);
  float*  kf    = (float*)alloc((size_t)S_TOK * DIM * 4);
  float*  vf    = (float*)alloc((size_t)S_TOK * DIM * 4);
  bf16_t* kc16  = (bf16_t*)alloc((size_t)S_TOK * DCMP * 2);
  bf16_t* vc16  = (bf16_t*)alloc((size_t)S_TOK * DCMP * 2);
  bf16_t* qT    = (bf16_t*)alloc((size_t)S_TOK * DIM * 2);
  bf16_t* kT    = (bf16_t*)alloc((size_t)S_TOK * DIM * 2);
  bf16_t* vT    = (bf16_t*)alloc((size_t)S_TOK * DIM * 2);
  bf16_t* sa16  = (bf16_t*)alloc((size_t)S_TOK * DIM * 2);
  float*  sao   = (float*)alloc((size_t)S_TOK * DIM * 4);
  float*  x     = (float*)alloc((size_t)S_TOK * DIM * 4);
  bf16_t* x16   = (bf16_t*)alloc((size_t)S_TOK * DIM * 2);
  float*  mid   = (float*)alloc((size_t)S_TOK * 2 * HIDN * 4);
  bf16_t* act16 = (bf16_t*)alloc((size_t)S_TOK * HIDN * 2);
  float*  shrd  = (float*)alloc((size_t)S_TOK * DIM * 4);
  float*  routed= (float*)alloc((size_t)S_TOK * DIM * 4);
  float*  wgate = (float*)alloc((size_t)S_TOK * NEXP * 4);

  auto cvt = [&](const float* in, bf16_t* o, size_t n) {
    f32_to_bf16_kernel<<<1024, 256, 0, stream>>>(in, o, n);
  };
  auto gemm_f32 = [&](const bf16_t* A, const bf16_t* B, float* C, int M, int N, int K) {
    gemm_bf16_kernel<false, false><<<dim3(N / GBN, M / GBM), 256, 0, stream>>>(
        A, B, (void*)C, nullptr, 0, M, N, K);
  };
  auto gemm_b16 = [&](const bf16_t* A, const bf16_t* B, bf16_t* C, int M, int N, int K) {
    gemm_bf16_kernel<true, false><<<dim3(N / GBN, M / GBM), 256, 0, stream>>>(
        A, B, (void*)C, nullptr, 0, M, N, K);
  };
  auto gemm_moe = [&](const bf16_t* A, const bf16_t* B, float* C,
                      const float* rs, int stride, int M, int N, int K) {
    gemm_bf16_kernel<false, true><<<dim3(N / GBN, M / GBM), 256, 0, stream>>>(
        A, B, (void*)C, rs, stride, M, N, K);
  };

  // 1) downconvert weights + activations to bf16
  cvt(Wq, Wq16, (size_t)DIM * DIM);
  cvt(Wk_c, Wkc16, (size_t)DIM * DCMP);
  cvt(Wv_c, Wvc16, (size_t)DIM * DCMP);
  cvt(Wk, Wk16, (size_t)DCMP * DIM);
  cvt(Wv, Wv16, (size_t)DCMP * DIM);
  cvt(Wo, Wo16, (size_t)DIM * DIM);
  cvt(Wsi, Wsi16, (size_t)DIM * 2 * HIDN);
  cvt(Wso, Wso16, (size_t)HIDN * DIM);
  cvt(W1, W116, (size_t)DIM * 2 * HIDN);
  cvt(W2, W216, (size_t)NEXP * HIDN * DIM);
  cvt(src, src16, (size_t)S_TOK * DIM);

  // 2) QKV projections (low-rank path for K/V)
  gemm_f32(src16, Wq16, qf, S_TOK, DIM, DIM);
  gemm_b16(src16, Wkc16, kc16, S_TOK, DCMP, DIM);
  gemm_b16(src16, Wvc16, vc16, S_TOK, DCMP, DIM);
  gemm_f32(kc16, Wk16, kf, S_TOK, DIM, DCMP);
  gemm_f32(vc16, Wv16, vf, S_TOK, DIM, DCMP);

  // 3) RoPE + transpose to [B][H][T][64] bf16
  rope_qkv_kernel<<<BB * TT * NH, 32, 0, stream>>>(qf, kf, vf, qT, kT, vT);

  // 4) WMMA flash attention -> sa16 [B][T][D]
  flash_attn_kernel<<<dim3(TT / 64, BB * NH), 128, 0, stream>>>(qT, kT, vT, sa16);

  // 5) output projection + residual + rmsnorm1
  gemm_f32(sa16, Wo16, sao, S_TOK, DIM, DIM);
  add_rmsnorm_kernel<<<S_TOK, 256, 0, stream>>>(src, sao, nullptr, n1w, x, x16);

  // 6) gating (fp32, tiny)
  gate_kernel<<<S_TOK, 128, 0, stream>>>(x, Wg, We, gbias, ebias, wgate);

  // 7) shared expert: swiglu(x @ Wsi) @ Wso
  gemm_f32(x16, Wsi16, mid, S_TOK, 2 * HIDN, DIM);
  swiglu_kernel<<<2048, 256, 0, stream>>>(mid, act16, (size_t)S_TOK * HIDN);
  gemm_f32(act16, Wso16, shrd, S_TOK, DIM, HIDN);

  // 8) routed experts: h = swiglu(x @ W1); routed += (h @ W2[e]) * w[:,e]
  gemm_f32(x16, W116, mid, S_TOK, 2 * HIDN, DIM);
  swiglu_kernel<<<2048, 256, 0, stream>>>(mid, act16, (size_t)S_TOK * HIDN);
  fill_zero_kernel<<<1024, 256, 0, stream>>>(routed, (size_t)S_TOK * DIM);
  for (int e = 0; e < NEXP; ++e) {
    gemm_moe(act16, W216 + (size_t)e * HIDN * DIM, routed,
             wgate + e, NEXP, S_TOK, DIM, HIDN);
  }

  // 9) final residual + rmsnorm2 -> d_out (fp32)
  add_rmsnorm_kernel<<<S_TOK, 256, 0, stream>>>(x, shrd, routed, n2w, out, nullptr);
}
